// MegaBlock_48988396978449
// MI455X (gfx1250) — matrix operand order repaired and checked by reading, not compiled, not run on hardware
//
#include <hip/hip_runtime.h>
#include <hip/hip_bf16.h>
#include <cstdint>

// ---------------------------------------------------------------------------
// MegaBlock forward for MI455X (gfx1250): bf16 WMMA GEMMs + recurrence EMA +
// fused chunk attention. Wave32, 8 waves / 256-thread blocks.
// Async global->LDS staging (ASYNCcnt) with double-buffered K-loop.
// ---------------------------------------------------------------------------

#define BB 4
#define LL 4096
#define DD 1024
#define ZZ 128
#define VV 1024
#define NN 16
#define CC 128
#define BL (BB * LL)             // 16384
#define NCOLS (ZZ + VV + 2 * DD) // 3200

typedef __attribute__((ext_vector_type(16))) __bf16 v16bf;
typedef __attribute__((ext_vector_type(8)))  float  v8f;

#if defined(__HIP_DEVICE_COMPILE__) && __has_builtin(__builtin_amdgcn_global_load_async_to_lds_b128)
#define HAVE_ASYNC_LDS 1
#else
#define HAVE_ASYNC_LDS 0
#endif

#if HAVE_ASYNC_LDS
#if __has_builtin(__builtin_amdgcn_s_wait_asynccnt)
#define WAIT_ASYNC(n) __builtin_amdgcn_s_wait_asynccnt(n)
#else
#define WAIT_ASYNC(n) asm volatile("s_wait_asynccnt %0" ::"n"(n) : "memory")
#endif
#else
#define WAIT_ASYNC(n) do {} while (0)
#endif

// ---------------- helpers -------------------------------------------------

__device__ __forceinline__ unsigned short f2bf_us(float f) {
  unsigned int u = __float_as_uint(f);
  u += 0x7fffu + ((u >> 16) & 1u);   // round-to-nearest-even
  return (unsigned short)(u >> 16);
}
__device__ __forceinline__ __bf16 f2bf(float f) {
  union { unsigned short u; __bf16 b; } c;
  c.u = f2bf_us(f);
  return c.b;
}
__device__ __forceinline__ float sigmoidf_(float x) { return 1.f / (1.f + __expf(-x)); }
__device__ __forceinline__ float siluf_(float x)    { return x / (1.f + __expf(-x)); }

// Stage 16 bytes global -> LDS. Async (ASYNCcnt-tracked) when available.
// Builtin signature (per hipcc diagnostic): param0 = addrspace(1) pointer to
// 'int __attribute__((vector_size(16)))'; LDS destination in addrspace(3).
__device__ __forceinline__ void stage16(const __bf16* g, __bf16* l) {
#if HAVE_ASYNC_LDS
  typedef int v4i_ __attribute__((vector_size(16)));
  typedef __attribute__((address_space(1))) v4i_ GT;
  typedef __attribute__((address_space(3))) v4i_ LT;
  __builtin_amdgcn_global_load_async_to_lds_b128(
      (GT*)(uintptr_t)g, (LT*)(uint32_t)(uintptr_t)l, 0, 0);
#else
  *(uint4*)l = *(const uint4*)g;
#endif
}

__device__ __forceinline__ v8f wmma_bf16(v16bf a, v16bf b, v8f c) {
  // v_wmma_f32_16x16x32_bf16  D = A(16x32) * B(32x16) + C
  return __builtin_amdgcn_wmma_f32_16x16x32_bf16(false, a, false, b, (short)0, c, false, false);
}

// A fragment: 16x32 bf16, rows m0..m0+15, cols k0..k0+31, from row-major LDS.
// Layout: lanes 0-15 row=lane, K 0-15;
// lanes 16-31 same rows, K 16-31 (matches load_frag_b).
__device__ __forceinline__ v16bf load_frag_a(const __bf16* A, int lda, int m0, int k0, int lane) {
  const int row = m0 + (lane & 15);
  const int kh  = (lane >> 4) * 16;
  v16bf f;
#pragma unroll
  for (int v = 0; v < 8; ++v) {
    const int kk = k0 + kh + v * 2;
    f[2 * v]     = A[row * lda + kk];
    f[2 * v + 1] = A[row * lda + kk + 1];
  }
  return f;
}

// B fragment: 32x16 bf16 (KxN) from row-major (K-major) LDS tile.
// Lanes 0-15: N=lane, K 0-15; lanes 16-31: same N, K 16-31.
__device__ __forceinline__ v16bf load_frag_b(const __bf16* Bm, int ldb, int k0, int n0, int lane) {
  const int col = n0 + (lane & 15);
  const int kb  = k0 + ((lane >> 4) << 4);
  v16bf f;
#pragma unroll
  for (int v = 0; v < 8; ++v) {
    f[2 * v]     = Bm[(kb + 2 * v) * ldb + col];
    f[2 * v + 1] = Bm[(kb + 2 * v + 1) * ldb + col];
  }
  return f;
}

// B fragment where B is stored transposed (N-major): B[k][n] = Bm[n*ld + k].
__device__ __forceinline__ v16bf load_frag_bT(const __bf16* Bm, int ld, int k0, int n0, int lane) {
  const int col = n0 + (lane & 15);
  const int kb  = k0 + ((lane >> 4) << 4);
  v16bf f;
#pragma unroll
  for (int v = 0; v < 8; ++v) {
    f[2 * v]     = Bm[col * ld + kb + 2 * v];
    f[2 * v + 1] = Bm[col * ld + kb + 2 * v + 1];
  }
  return f;
}

// ---------------- kernels -------------------------------------------------

__global__ __launch_bounds__(256) void cvt_f32_bf16_kernel(const float* __restrict__ s,
                                                           __bf16* __restrict__ d, int n) {
  int i = blockIdx.x * 256 + threadIdx.x;
  if (i < n) d[i] = f2bf(s[i]);
}

// LayerNorm over D=1024, one row per block; writes fp32 + bf16 copies.
__global__ __launch_bounds__(256) void ln1_kernel(const float* __restrict__ x,
                                                  const float* __restrict__ g,
                                                  const float* __restrict__ bb,
                                                  float* __restrict__ xn,
                                                  __bf16* __restrict__ xnb) {
  const size_t row = blockIdx.x;
  const int t = threadIdx.x;
  __shared__ float r0[256], r1[256];
  float loc[4], s = 0.f, s2 = 0.f;
#pragma unroll
  for (int i = 0; i < 4; ++i) {
    float v = x[row * DD + t + i * 256];
    loc[i] = v; s += v; s2 += v * v;
  }
  r0[t] = s; r1[t] = s2; __syncthreads();
  for (int off = 128; off > 0; off >>= 1) {
    if (t < off) { r0[t] += r0[t + off]; r1[t] += r1[t + off]; }
    __syncthreads();
  }
  const float mean = r0[0] * (1.f / DD);
  const float var  = r1[0] * (1.f / DD) - mean * mean;
  const float rs   = rsqrtf(var + 1e-5f);
#pragma unroll
  for (int i = 0; i < 4; ++i) {
    const int c = t + i * 256;
    const float v = (loc[i] - mean) * rs * g[c] + bb[c];
    xn[row * DD + c]  = v;
    xnb[row * DD + c] = f2bf(v);
  }
}

// Multi-head EMA as a 16-state linear recurrence per (b,d) channel.
// mx = silu(conv(xn,k) + xn*omega), stored bf16 for the Wmx GEMM.
__global__ __launch_bounds__(256) void ema_kernel(const float* __restrict__ xn,
                                                  const float* __restrict__ delta,
                                                  const float* __restrict__ alpha,
                                                  const float* __restrict__ beta_e,
                                                  const float* __restrict__ gamma_e,
                                                  const float* __restrict__ omega,
                                                  __bf16* __restrict__ mxb) {
  const int idx = blockIdx.x * 256 + threadIdx.x;   // 0 .. B*D-1
  const int b = idx >> 10;
  const int d = idx & (DD - 1);
  float q[NN], w[NN], s[NN];
  const float om = omega[d];
#pragma unroll
  for (int n = 0; n < NN; ++n) {
    const float p = sigmoidf_(delta[d * NN + n]);
    const float a = sigmoidf_(alpha[d * NN + n]);
    q[n] = 1.f - p * a;
    w[n] = p * beta_e[d * NN + n] * gamma_e[d * NN + n] * 0.25f;  // (1/N)^0.5
    s[n] = 0.f;
  }
  const float* xp = xn + (size_t)b * LL * DD + d;
  __bf16* op = mxb + (size_t)b * LL * DD + d;
  for (int l = 0; l < LL; ++l) {
    const float xv = xp[(size_t)l * DD];
    float y = om * xv;
#pragma unroll
    for (int n = 0; n < NN; ++n) { s[n] = q[n] * s[n] + xv; y += w[n] * s[n]; }
    op[(size_t)l * DD] = f2bf(siluf_(y));
  }
}

// Generic bf16 WMMA GEMM: out = ACT(A@B + bias + addsrc). 128x128 tile, BK=64,
// double-buffered async LDS staging; 8 wave32 (4x2), 32x64 per wave,
// 16 v_wmma_f32_16x16x32_bf16 per wave per stage.
template <int ACT>
__global__ __launch_bounds__(256) void gemm_bf16_wmma(const __bf16* __restrict__ A,
                                                      const __bf16* __restrict__ Bw,
                                                      const float* __restrict__ bias,
                                                      const float* __restrict__ addsrc,
                                                      float* __restrict__ outF,
                                                      __bf16* __restrict__ outB,
                                                      int M, int N, int K) {
  extern __shared__ char smem[];
  // layout per buffer (32KB stride): A 128x64 (16KB) then B 64x128 (16KB).
  // NOTE: no LDS pointer arrays (lld rejects addrspacecast in static inits);
  // buffer addresses are computed arithmetically from smem at each use.
  const int t = threadIdx.x, lane = t & 31, wave = t >> 5;
  const int wm = wave >> 1, wn = wave & 1;
  const int m_blk = blockIdx.y * 128, n_blk = blockIdx.x * 128;

  v8f acc[2][4];
#pragma unroll
  for (int i = 0; i < 2; ++i)
#pragma unroll
    for (int j = 0; j < 4; ++j) acc[i][j] = (v8f){};

  const int a_row = t >> 1, a_seg = (t & 1) * 32;   // 128 rows x 64 cols
  const int b_row = t >> 2, b_seg = (t & 3) * 32;   // 64 rows x 128 cols

  auto issue_tile = [&](int k0, int buf) {
    char* base = smem + (buf & 1) * 32768;
    const __bf16* ga = A + (size_t)(m_blk + a_row) * K + k0 + a_seg;
    __bf16* sa = (__bf16*)base + a_row * 64 + a_seg;
    const __bf16* gb = Bw + (size_t)(k0 + b_row) * N + n_blk + b_seg;
    __bf16* sb = (__bf16*)(base + 16384) + b_row * 128 + b_seg;
#pragma unroll
    for (int i = 0; i < 4; ++i) {   // 4 x 16B = 64B each for A and B
      stage16(ga + i * 8, sa + i * 8);
      stage16(gb + i * 8, sb + i * 8);
    }
    if (k0 + 64 < K) {   // global_prefetch_b8 of the tile after next
      __builtin_prefetch(ga + 64, 0, 0);
      __builtin_prefetch(gb + (size_t)64 * N, 0, 0);
    }
  };

  const int nk = K >> 6;
  issue_tile(0, 0);
  for (int i = 0; i < nk; ++i) {
    if (i + 1 < nk) { issue_tile((i + 1) << 6, (i + 1) & 1); WAIT_ASYNC(8); }
    else            { WAIT_ASYNC(0); }
    __syncthreads();
    const __bf16* As = (const __bf16*)(smem + (i & 1) * 32768);
    const __bf16* Bs = (const __bf16*)(smem + (i & 1) * 32768 + 16384);
#pragma unroll
    for (int kk = 0; kk < 2; ++kk) {
      v16bf af0 = load_frag_a(As, 64, wm * 32, kk * 32, lane);
      v16bf af1 = load_frag_a(As, 64, wm * 32 + 16, kk * 32, lane);
#pragma unroll
      for (int j = 0; j < 4; ++j) {
        v16bf bf = load_frag_b(Bs, 128, kk * 32, wn * 64 + j * 16, lane);
        acc[0][j] = wmma_bf16(af0, bf, acc[0][j]);
        acc[1][j] = wmma_bf16(af1, bf, acc[1][j]);
      }
    }
    __syncthreads();   // buffer (i&1) free for reuse at stage i+2
  }

#pragma unroll
  for (int i = 0; i < 2; ++i)
#pragma unroll
    for (int j = 0; j < 4; ++j) {
      const int n  = n_blk + wn * 64 + j * 16 + (lane & 15);
      const int mb = m_blk + wm * 32 + i * 16 + ((lane >> 4) << 3);
      const float bvv = bias ? bias[n] : 0.f;
#pragma unroll
      for (int r = 0; r < 8; ++r) {
        const size_t m = (size_t)(mb + r);
        float v = acc[i][j][r] + bvv;
        if (addsrc) v += addsrc[m * N + n];
        if (ACT == 1) v = siluf_(v);
        if (outF) outF[m * N + n] = v;
        if (outB) outB[m * N + n] = f2bf(v);
      }
    }
}

// Split base (BLx3200) -> u (sigmoid), q/k (silu->affine->bf16), r (silu), hx.
__global__ __launch_bounds__(256) void split_base_kernel(const float* __restrict__ base,
                                                         const float* __restrict__ gamma_g,
                                                         const float* __restrict__ beta_g,
                                                         float* __restrict__ u,
                                                         __bf16* __restrict__ qb,
                                                         __bf16* __restrict__ kb,
                                                         float* __restrict__ rg,
                                                         float* __restrict__ hx) {
  const size_t row = blockIdx.x;
  const float* br = base + row * NCOLS;
  const int t = threadIdx.x;
#pragma unroll
  for (int i = 0; i < 4; ++i) {
    const int c = t + i * 256;
    u[row * DD + c] = sigmoidf_(br[c]);
    const float rv = siluf_(br[DD + ZZ + c]);
    rg[row * VV + c] = rv;
    hx[row * DD + c] = br[DD + ZZ + VV + c];
  }
  if (t < ZZ) {
    const float zv = siluf_(br[DD + t]);
    const float qv = (zv * gamma_g[t] + beta_g[t]) * 0.088388347648318447f; // Z^-1/2
    const float kv = zv * gamma_g[ZZ + t] + beta_g[ZZ + t];
    qb[row * ZZ + t] = f2bf(qv);
    kb[row * ZZ + t] = f2bf(kv);
  }
}

// Fused chunk attention: one block per (b,chunk). S = QK^T + rpb bias, softmax,
// H = A@V with r-gate fused into bf16 store (input of Wh GEMM). V tiles are
// async-pipelined, ping-ponging between the dead K region (offset 32KB) and
// the dead score region (offset 64KB).
__global__ __launch_bounds__(256) void attn_kernel(const __bf16* __restrict__ qb,
                                                   const __bf16* __restrict__ kb,
                                                   const __bf16* __restrict__ vbf,
                                                   const float* __restrict__ rpb,
                                                   const float* __restrict__ rg,
                                                   __bf16* __restrict__ hrb) {
  extern __shared__ char smem[];
  __bf16* Qs = (__bf16*)smem;              // 128x128 bf16 (later reused for attn)
  __bf16* Ks = (__bf16*)(smem + 32768);    // 128x128 bf16 (later V tile buf0)
  float*  Ss = (float*)(smem + 65536);     // 128x128 f32 scores (later V buf1)
  const int t = threadIdx.x, lane = t & 31, wave = t >> 5;
  const size_t rowbase = (size_t)blockIdx.x * CC;  // (b*32+kc)*128 == b*L + kc*C
  const int row = t >> 1, hs = (t & 1) * 64;

  { // stage Q, K (async)
    const __bf16* gq = qb + (rowbase + row) * ZZ + hs;
    const __bf16* gk = kb + (rowbase + row) * ZZ + hs;
    __bf16* sq = Qs + row * ZZ + hs;
    __bf16* sk = Ks + row * ZZ + hs;
#pragma unroll
    for (int i = 0; i < 8; ++i) { stage16(gq + i * 8, sq + i * 8); stage16(gk + i * 8, sk + i * 8); }
  }
  WAIT_ASYNC(0);
  __syncthreads();

  // S = Q @ K^T + bias;  wave handles rows 16w..16w+15
  for (int nt = 0; nt < 8; ++nt) {
    v8f acc = (v8f){};
#pragma unroll
    for (int kk = 0; kk < 4; ++kk) {
      v16bf a = load_frag_a(Qs, ZZ, wave * 16, kk * 32, lane);
      v16bf b = load_frag_bT(Ks, ZZ, kk * 32, nt * 16, lane);
      acc = wmma_bf16(a, b, acc);
    }
    const int n  = nt * 16 + (lane & 15);
    const int mb = wave * 16 + ((lane >> 4) << 3);
#pragma unroll
    for (int r = 0; r < 8; ++r) {
      const int m = mb + r;
      Ss[m * CC + n] = acc[r] + rpb[CC - 1 + n - m];
    }
  }
  __syncthreads();   // Ks (K matrix) now dead

  // V buffer vt: even -> smem+32KB (old K), odd -> smem+64KB (old scores)
  auto issueV = [&](int vt) {
    const __bf16* gv = vbf + (rowbase + row) * VV + vt * 128 + hs;
    __bf16* sv = (__bf16*)(smem + 32768 + (vt & 1) * 32768) + row * 128 + hs;
#pragma unroll
    for (int i = 0; i < 8; ++i) stage16(gv + i * 8, sv + i * 8);
  };
  issueV(0);   // overlaps softmax

  // row softmax -> bf16 attn into Qs
  if (t < CC) {
    float mx = -1e30f;
    for (int j = 0; j < CC; ++j) mx = fmaxf(mx, Ss[t * CC + j]);
    float sum = 0.f;
    for (int j = 0; j < CC; ++j) { const float e = __expf(Ss[t * CC + j] - mx); Ss[t * CC + j] = e; sum += e; }
    const float inv = 1.f / sum;
    for (int j = 0; j < CC; ++j) Qs[t * CC + j] = f2bf(Ss[t * CC + j] * inv);
  }
  __syncthreads();   // Ss (scores) now dead

  v16bf af[4];
#pragma unroll
  for (int kk = 0; kk < 4; ++kk) af[kk] = load_frag_a(Qs, CC, wave * 16, kk * 32, lane);
  issueV(1);

  for (int vt = 0; vt < 8; ++vt) {
    if (vt < 7) { WAIT_ASYNC(8); } else { WAIT_ASYNC(0); }
    __syncthreads();
    const __bf16* Vt = (const __bf16*)(smem + 32768 + (vt & 1) * 32768);
    for (int nt = 0; nt < 8; ++nt) {
      v8f acc = (v8f){};
#pragma unroll
      for (int kk = 0; kk < 4; ++kk) {
        v16bf b = load_frag_b(Vt, 128, kk * 32, nt * 16, lane);
        acc = wmma_bf16(af[kk], b, acc);
      }
      const int n  = nt * 16 + (lane & 15);
      const int mb = wave * 16 + ((lane >> 4) << 3);
#pragma unroll
      for (int r = 0; r < 8; ++r) {
        const size_t grow = rowbase + mb + r;
        const size_t gcol = (size_t)vt * 128 + n;
        hrb[grow * VV + gcol] = f2bf(acc[r] * rg[grow * VV + gcol]);
      }
    }
    __syncthreads();               // buffer (vt&1) free for reuse
    if (vt + 2 < 8) issueV(vt + 2);
  }
}

// out = residual + u*(h2 - residual); LN2(out) -> bf16 for the MLP GEMM.
__global__ __launch_bounds__(256) void out_ln2_kernel(const float* __restrict__ x,
                                                      const float* __restrict__ u,
                                                      const float* __restrict__ h2,
                                                      const float* __restrict__ g,
                                                      const float* __restrict__ bb,
                                                      float* __restrict__ outb,
                                                      __bf16* __restrict__ yb) {
  const size_t row = blockIdx.x;
  const int t = threadIdx.x;
  __shared__ float r0[256], r1[256];
  float loc[4], s = 0.f, s2 = 0.f;
#pragma unroll
  for (int i = 0; i < 4; ++i) {
    const int c = t + i * 256;
    const float res = x[row * DD + c];
    const float o = res + u[row * DD + c] * (h2[row * DD + c] - res);
    loc[i] = o;
    outb[row * DD + c] = o;
    s += o; s2 += o * o;
  }
  r0[t] = s; r1[t] = s2; __syncthreads();
  for (int off = 128; off > 0; off >>= 1) {
    if (t < off) { r0[t] += r0[t + off]; r1[t] += r1[t + off]; }
    __syncthreads();
  }
  const float mean = r0[0] * (1.f / DD);
  const float var  = r1[0] * (1.f / DD) - mean * mean;
  const float rs   = rsqrtf(var + 1e-5f);
#pragma unroll
  for (int i = 0; i < 4; ++i) {
    const int c = t + i * 256;
    yb[row * DD + c] = f2bf((loc[i] - mean) * rs * g[c] + bb[c]);
  }
}

// ---------------- host launcher -------------------------------------------

extern "C" void kernel_launch(void* const* d_in, const int* in_sizes, int n_in,
                              void* d_out, int out_size, void* d_ws, size_t ws_size,
                              hipStream_t stream) {
  (void)in_sizes; (void)n_in; (void)out_size; (void)ws_size;
  const float* x      = (const float*)d_in[0];
  const float* ln1_g  = (const float*)d_in[1];
  const float* ln1_b  = (const float*)d_in[2];
  const float* delta  = (const float*)d_in[3];
  const float* alpha  = (const float*)d_in[4];
  const float* beta_e = (const float*)d_in[5];
  const float* gamma_e= (const float*)d_in[6];
  const float* omega  = (const float*)d_in[7];
  const float* Wv     = (const float*)d_in[8];
  const float* bv     = (const float*)d_in[9];
  const float* Wmx    = (const float*)d_in[10];
  const float* bmx    = (const float*)d_in[11];
  const float* Wh     = (const float*)d_in[12];
  const float* bh     = (const float*)d_in[13];
  const float* gamma_g= (const float*)d_in[14];
  const float* beta_g = (const float*)d_in[15];
  const float* rpb    = (const float*)d_in[16];
  const float* ln2_g  = (const float*)d_in[17];
  const float* ln2_b  = (const float*)d_in[18];
  const float* W1     = (const float*)d_in[19];
  const float* b1     = (const float*)d_in[20];
  const float* W2     = (const float*)d_in[21];
  const float* b2     = (const float*)d_in[22];
  float* out = (float*)d_out;

  char* w = (char*)d_ws;
  size_t off = 0;
  auto carve = [&](size_t bytes) -> void* {
    void* p = w + off;
    off = (off + bytes + 255) & ~(size_t)255;
    return p;
  };
  float*  xn   = (float*)carve((size_t)BL * DD * 4);
  __bf16* xnb  = (__bf16*)carve((size_t)BL * DD * 2);
  __bf16* mxb  = (__bf16*)carve((size_t)BL * DD * 2);
  __bf16* vbf  = (__bf16*)carve((size_t)BL * VV * 2);
  float*  base = (float*)carve((size_t)BL * NCOLS * 4);
  float*  u    = (float*)carve((size_t)BL * DD * 4);
  __bf16* qb   = (__bf16*)carve((size_t)BL * ZZ * 2);
  __bf16* kb   = (__bf16*)carve((size_t)BL * ZZ * 2);
  float*  rg   = (float*)carve((size_t)BL * VV * 4);
  float*  hx   = (float*)carve((size_t)BL * DD * 4);
  __bf16* hrb  = (__bf16*)carve((size_t)BL * VV * 2);
  float*  h2   = (float*)carve((size_t)BL * DD * 4);
  float*  outb = (float*)carve((size_t)BL * DD * 4);
  __bf16* yb   = (__bf16*)carve((size_t)BL * DD * 2);
  __bf16* tb   = (__bf16*)carve((size_t)BL * 2 * DD * 2);
  __bf16* Wvb  = (__bf16*)carve((size_t)DD * VV * 2);
  __bf16* Wmxb = (__bf16*)carve((size_t)DD * NCOLS * 2);
  __bf16* Whb  = (__bf16*)carve((size_t)VV * DD * 2);
  __bf16* W1b  = (__bf16*)carve((size_t)DD * 2 * DD * 2);
  __bf16* W2b  = (__bf16*)carve((size_t)2 * DD * DD * 2);

  (void)hipFuncSetAttribute(reinterpret_cast<const void*>(attn_kernel),
                            hipFuncAttributeMaxDynamicSharedMemorySize, 131072);
  (void)hipFuncSetAttribute(reinterpret_cast<const void*>(gemm_bf16_wmma<0>),
                            hipFuncAttributeMaxDynamicSharedMemorySize, 65536);
  (void)hipFuncSetAttribute(reinterpret_cast<const void*>(gemm_bf16_wmma<1>),
                            hipFuncAttributeMaxDynamicSharedMemorySize, 65536);

  // weight conversions f32 -> bf16
  cvt_f32_bf16_kernel<<<(DD * VV + 255) / 256, 256, 0, stream>>>(Wv, Wvb, DD * VV);
  cvt_f32_bf16_kernel<<<(DD * NCOLS + 255) / 256, 256, 0, stream>>>(Wmx, Wmxb, DD * NCOLS);
  cvt_f32_bf16_kernel<<<(VV * DD + 255) / 256, 256, 0, stream>>>(Wh, Whb, VV * DD);
  cvt_f32_bf16_kernel<<<(DD * 2 * DD + 255) / 256, 256, 0, stream>>>(W1, W1b, DD * 2 * DD);
  cvt_f32_bf16_kernel<<<(2 * DD * DD + 255) / 256, 256, 0, stream>>>(W2, W2b, 2 * DD * DD);

  // LN1
  ln1_kernel<<<BL, 256, 0, stream>>>(x, ln1_g, ln1_b, xn, xnb);
  // EMA (16-state recurrence) -> silu -> mx (bf16)
  ema_kernel<<<(BB * DD) / 256, 256, 0, stream>>>(xn, delta, alpha, beta_e, gamma_e, omega, mxb);
  // v = silu(xn @ Wv + bv)  (bf16 out for attention A@V)
  gemm_bf16_wmma<1><<<dim3(VV / 128, BL / 128), 256, 65536, stream>>>(
      xnb, Wvb, bv, nullptr, nullptr, vbf, BL, VV, DD);
  // base = mx @ Wmx + bmx (raw f32)
  gemm_bf16_wmma<0><<<dim3(NCOLS / 128, BL / 128), 256, 65536, stream>>>(
      mxb, Wmxb, bmx, nullptr, base, nullptr, BL, NCOLS, DD);
  // split into u, q, k, r, hx
  split_base_kernel<<<BL, 256, 0, stream>>>(base, gamma_g, beta_g, u, qb, kb, rg, hx);
  // chunk attention; stores bf16(h*r)
  attn_kernel<<<BB * (LL / CC), 256, 131072, stream>>>(qb, kb, vbf, rpb, rg, hrb);
  // h2 = silu(hx + (h*r) @ Wh + bh)
  gemm_bf16_wmma<1><<<dim3(DD / 128, BL / 128), 256, 65536, stream>>>(
      hrb, Whb, bh, hx, h2, nullptr, BL, DD, VV);
  // out = res + u*(h2-res); yb = bf16(LN2(out))
  out_ln2_kernel<<<BL, 256, 0, stream>>>(x, u, h2, ln2_g, ln2_b, outb, yb);
  // t = silu(y @ W1 + b1)
  gemm_bf16_wmma<1><<<dim3((2 * DD) / 128, BL / 128), 256, 65536, stream>>>(
      yb, W1b, b1, nullptr, nullptr, tb, BL, 2 * DD, DD);
  // final = out + (t @ W2 + b2)
  gemm_bf16_wmma<0><<<dim3(DD / 128, BL / 128), 256, 65536, stream>>>(
      tb, W2b, b2, outb, out, nullptr, BL, DD, 2 * DD);
}
